// TransformerBlock_48404281425975
// MI455X (gfx1250) — matrix operand order repaired and checked by reading, not compiled, not run on hardware
//
#include <hip/hip_runtime.h>

// ---------------- problem constants ----------------
#define B_SZ     2
#define T_SZ     2048
#define C_DIM    1024
#define HEADS    16
#define DHEAD    64
#define INNER    1024      // HEADS*DHEAD
#define MLP_DIM  4096
#define BT       (B_SZ*T_SZ)   // 4096 rows
#define QKV_W    (3*INNER)     // 3072

// ---------------- WMMA types (CDNA5, wave32) ----------------
typedef __attribute__((ext_vector_type(16))) __bf16 v16bf;
typedef __attribute__((ext_vector_type(8)))  float  v8f;

union Frag16 {
  v16bf          v;
  unsigned int   d[8];
  unsigned short u[16];
};

__device__ __forceinline__ unsigned short f32_to_bf16(float f) {
  unsigned int u = __float_as_uint(f);
  u += 0x7FFFu + ((u >> 16) & 1u);          // round-to-nearest-even
  return (unsigned short)(u >> 16);
}

// A-matrix 16x32 bf16 per-lane K map: dword j covers K = kmapA, kmapA+1
__device__ __forceinline__ int kmapA(int j, int khalf) {
  return (j < 4 ? 2 * j : 16 + 2 * (j - 4)) + 8 * khalf;
}
// B-matrix 32x16 bf16 per-lane K map: same K order as kmapA
__device__ __forceinline__ int kmapB(int j, int khalf) {
  return (j < 4 ? 2 * j : 16 + 2 * (j - 4)) + 8 * khalf;
}

__device__ __forceinline__ v8f wmma_bf16(const Frag16& a, const Frag16& b, v8f c) {
  return __builtin_amdgcn_wmma_f32_16x16x32_bf16(
      false, a.v, false, b.v, (short)0, c, false, false);
}

__device__ __forceinline__ float red_max16(float v) {
  v = fmaxf(v, __shfl_xor(v, 1, 32));
  v = fmaxf(v, __shfl_xor(v, 2, 32));
  v = fmaxf(v, __shfl_xor(v, 4, 32));
  v = fmaxf(v, __shfl_xor(v, 8, 32));
  return v;
}
__device__ __forceinline__ float red_sum16(float v) {
  v += __shfl_xor(v, 1, 32);
  v += __shfl_xor(v, 2, 32);
  v += __shfl_xor(v, 4, 32);
  v += __shfl_xor(v, 8, 32);
  return v;
}

// ---------------- fp32 -> bf16 convert ----------------
__global__ __launch_bounds__(256)
void cvt_bf16_kernel(const float* __restrict__ in, unsigned short* __restrict__ out, int n) {
  int i = blockIdx.x * blockDim.x + threadIdx.x;
  int stride = gridDim.x * blockDim.x;
  for (; i < n; i += stride) out[i] = f32_to_bf16(in[i]);
}

// ---------------- LayerNorm (row of 1024) -> bf16 ----------------
__global__ __launch_bounds__(256)
void layernorm_bf16_kernel(const float* __restrict__ x, const float* __restrict__ g,
                           const float* __restrict__ b, unsigned short* __restrict__ out) {
  const int row = blockIdx.x;
  const int tid = threadIdx.x;
  const float* xr = x + (size_t)row * C_DIM;
  __shared__ float sm[256];

  float v[4];
  float s = 0.f;
#pragma unroll
  for (int i = 0; i < 4; ++i) { v[i] = xr[tid + i * 256]; s += v[i]; }
  sm[tid] = s; __syncthreads();
  for (int st = 128; st > 0; st >>= 1) { if (tid < st) sm[tid] += sm[tid + st]; __syncthreads(); }
  const float mu = sm[0] * (1.0f / C_DIM);
  __syncthreads();

  float s2 = 0.f;
#pragma unroll
  for (int i = 0; i < 4; ++i) { float d = v[i] - mu; s2 += d * d; }
  sm[tid] = s2; __syncthreads();
  for (int st = 128; st > 0; st >>= 1) { if (tid < st) sm[tid] += sm[tid + st]; __syncthreads(); }
  const float rs = rsqrtf(sm[0] * (1.0f / C_DIM) + 1e-5f);

#pragma unroll
  for (int i = 0; i < 4; ++i) {
    int c = tid + i * 256;
    out[(size_t)row * C_DIM + c] = f32_to_bf16((v[i] - mu) * rs * g[c] + b[c]);
  }
}

// ---------------- WMMA GEMM: C(MxN) = A(MxK,bf16) * B(KxN,bf16) ----------------
// EPI 0: store bf16.  EPI 1: fp32 = acc + bias + residual.  EPI 2: bf16 = gelu(acc+bias).
template <int EPI>
__global__ __launch_bounds__(256)
void gemm_bf16_kernel(const unsigned short* __restrict__ A, const unsigned short* __restrict__ B,
                      int M, int N, int K,
                      const float* __restrict__ bias, const float* __restrict__ res,
                      float* __restrict__ Cf, unsigned short* __restrict__ Cb) {
  __shared__ unsigned short As[128 * 32];   // [M][K]
  __shared__ unsigned short Bs[32 * 128];   // [K][N]

  const int tid  = threadIdx.x;
  const int wave = tid >> 5, lane = tid & 31;
  const int wm = wave >> 2, wn = wave & 3;          // 2x4 wave grid
  const int l16 = lane & 15, khalf = lane >> 4;
  const int bm = blockIdx.y * 128, bn = blockIdx.x * 128;

  v8f acc[4][2];
#pragma unroll
  for (int i = 0; i < 4; ++i)
#pragma unroll
    for (int j = 0; j < 2; ++j) acc[i][j] = {};

  for (int k0 = 0; k0 < K; k0 += 32) {
#pragma unroll
    for (int i = 0; i < 2; ++i) {                  // A tile 128x32
      int c = tid + i * 256;
      int r = c >> 2, col = (c & 3) * 8;
      *reinterpret_cast<uint4*>(&As[r * 32 + col]) =
        *reinterpret_cast<const uint4*>(&A[(size_t)(bm + r) * K + k0 + col]);
    }
#pragma unroll
    for (int i = 0; i < 2; ++i) {                  // B tile 32x128
      int c = tid + i * 256;
      int r = c >> 4, col = (c & 15) * 8;
      *reinterpret_cast<uint4*>(&Bs[r * 128 + col]) =
        *reinterpret_cast<const uint4*>(&B[(size_t)(k0 + r) * N + bn + col]);
    }
    // prefetch next K tiles into cache hierarchy (global_prefetch_b8)
    if (k0 + 32 < K) {
      __builtin_prefetch(&A[(size_t)(bm + (tid >> 1)) * K + (k0 + 32) + (tid & 1) * 16], 0, 1);
      __builtin_prefetch(&B[(size_t)(k0 + 32 + (tid >> 3)) * N + bn + (tid & 7) * 16], 0, 1);
    }
    __syncthreads();

    Frag16 bf[2];
#pragma unroll
    for (int nt = 0; nt < 2; ++nt) {
      int n = wn * 32 + nt * 16 + l16;
#pragma unroll
      for (int j = 0; j < 8; ++j) {
        int k = kmapB(j, khalf);
        bf[nt].u[2 * j]     = Bs[k * 128 + n];
        bf[nt].u[2 * j + 1] = Bs[(k + 1) * 128 + n];
      }
    }
#pragma unroll
    for (int mt = 0; mt < 4; ++mt) {
      Frag16 af;
      int m = wm * 64 + mt * 16 + l16;
#pragma unroll
      for (int j = 0; j < 8; ++j)
        af.d[j] = *reinterpret_cast<const unsigned int*>(&As[m * 32 + kmapA(j, khalf)]);
#pragma unroll
      for (int nt = 0; nt < 2; ++nt)
        acc[mt][nt] = wmma_bf16(af, bf[nt], acc[mt][nt]);
    }
    __syncthreads();
  }

#pragma unroll
  for (int mt = 0; mt < 4; ++mt)
#pragma unroll
    for (int nt = 0; nt < 2; ++nt) {
      int n_g = bn + wn * 32 + nt * 16 + l16;
#pragma unroll
      for (int j = 0; j < 8; ++j) {
        int m_g = bm + wm * 64 + mt * 16 + j + 8 * khalf;
        float v = acc[mt][nt][j];
        size_t idx = (size_t)m_g * N + n_g;
        if (EPI == 0) {
          Cb[idx] = f32_to_bf16(v);
        } else if (EPI == 1) {
          Cf[idx] = v + bias[n_g] + res[idx];
        } else {
          float t = v + bias[n_g];
          Cb[idx] = f32_to_bf16(0.5f * t * (1.0f + erff(t * 0.70710678118654752f)));
        }
      }
    }
}

// ---------------- causal flash attention: 64 queries x 1 head per block ----------------
__global__ __launch_bounds__(128)
void attention_kernel(const unsigned short* __restrict__ qkv, unsigned short* __restrict__ out) {
  const int h = blockIdx.y, b = blockIdx.z;
  const int q0 = blockIdx.x * 64;
  const int tid = threadIdx.x, wave = tid >> 5, lane = tid & 31;
  const int l16 = lane & 15, khalf = lane >> 4;
  const float scale = 0.125f;          // 1/sqrt(64)
  const float NEG_BIG = -1e30f;

  __shared__ unsigned short Qs[64 * 64];
  __shared__ unsigned short Ks[64 * 64];
  __shared__ unsigned short Vs[64 * 64];
  __shared__ unsigned short Ps[4][16 * 64];

  const size_t base_q = ((size_t)b * T_SZ + q0) * QKV_W + (size_t)h * DHEAD;

  // Q tile 64x64 -> LDS
#pragma unroll
  for (int i = 0; i < 4; ++i) {
    int c = tid + i * 128;
    int r = c >> 3, col = (c & 7) * 8;
    *reinterpret_cast<uint4*>(&Qs[r * 64 + col]) =
      *reinterpret_cast<const uint4*>(&qkv[base_q + (size_t)r * QKV_W + col]);
  }
  __syncthreads();

  const int qr = wave * 16;
  Frag16 aq[2];
#pragma unroll
  for (int f = 0; f < 2; ++f)
#pragma unroll
    for (int j = 0; j < 8; ++j)
      aq[f].d[j] = *reinterpret_cast<const unsigned int*>(
          &Qs[(qr + l16) * 64 + f * 32 + kmapA(j, khalf)]);

  float mrow[8], lrow[8];
  v8f oacc[4];
#pragma unroll
  for (int j = 0; j < 8; ++j) { mrow[j] = NEG_BIG; lrow[j] = 0.f; }
#pragma unroll
  for (int dt = 0; dt < 4; ++dt) oacc[dt] = {};

  const int nkb = q0 / 64 + 1;
  for (int kb = 0; kb < nkb; ++kb) {
    const size_t base_k = ((size_t)b * T_SZ + kb * 64) * QKV_W + INNER + (size_t)h * DHEAD;
    const size_t base_v = base_k + INNER;
#pragma unroll
    for (int i = 0; i < 4; ++i) {
      int c = tid + i * 128;
      int r = c >> 3, col = (c & 7) * 8;
      *reinterpret_cast<uint4*>(&Ks[r * 64 + col]) =
        *reinterpret_cast<const uint4*>(&qkv[base_k + (size_t)r * QKV_W + col]);
      *reinterpret_cast<uint4*>(&Vs[r * 64 + col]) =
        *reinterpret_cast<const uint4*>(&qkv[base_v + (size_t)r * QKV_W + col]);
    }
    // prefetch next key block's K/V rows (global_prefetch_b8)
    if (kb + 1 < nkb) {
      const size_t pbase = base_k + (size_t)64 * QKV_W;   // next kb, K plane
      int r = tid >> 1, half = (tid & 1) * 32;
      __builtin_prefetch(&qkv[pbase + (size_t)r * QKV_W + half], 0, 1);
      __builtin_prefetch(&qkv[pbase + INNER + (size_t)r * QKV_W + half], 0, 1);
    }
    __syncthreads();

    // S = Q @ K^T  (B[d][key] -> contiguous-d pairs from Ks[key][d])
    v8f s[4];
#pragma unroll
    for (int nt = 0; nt < 4; ++nt) {
      v8f sv = {};
      int key = nt * 16 + l16;
#pragma unroll
      for (int f = 0; f < 2; ++f) {
        Frag16 bk;
#pragma unroll
        for (int j = 0; j < 8; ++j)
          bk.d[j] = *reinterpret_cast<const unsigned int*>(
              &Ks[key * 64 + f * 32 + kmapB(j, khalf)]);
        sv = wmma_bf16(aq[f], bk, sv);
      }
      s[nt] = sv;
    }

    // scale + causal mask + online softmax
    float tmax[8];
#pragma unroll
    for (int j = 0; j < 8; ++j) tmax[j] = NEG_BIG;
#pragma unroll
    for (int nt = 0; nt < 4; ++nt) {
      int kg = kb * 64 + nt * 16 + l16;
#pragma unroll
      for (int j = 0; j < 8; ++j) {
        int qg = q0 + qr + j + 8 * khalf;
        float v = s[nt][j] * scale;
        if (kg > qg) v = NEG_BIG;
        s[nt][j] = v;
        tmax[j] = fmaxf(tmax[j], v);
      }
    }
    float mnew[8], fs[8], psum[8];
#pragma unroll
    for (int j = 0; j < 8; ++j) {
      mnew[j] = fmaxf(mrow[j], red_max16(tmax[j]));
      fs[j] = __expf(mrow[j] - mnew[j]);
      psum[j] = 0.f;
    }
#pragma unroll
    for (int nt = 0; nt < 4; ++nt)
#pragma unroll
      for (int j = 0; j < 8; ++j) {
        float p = __expf(s[nt][j] - mnew[j]);
        s[nt][j] = p;
        psum[j] += p;
      }
#pragma unroll
    for (int j = 0; j < 8; ++j) {
      lrow[j] = lrow[j] * fs[j] + red_sum16(psum[j]);
      mrow[j] = mnew[j];
    }
#pragma unroll
    for (int dt = 0; dt < 4; ++dt)
#pragma unroll
      for (int j = 0; j < 8; ++j) oacc[dt][j] *= fs[j];

    // P -> LDS bf16 (per-wave region), then O += P @ V
#pragma unroll
    for (int nt = 0; nt < 4; ++nt)
#pragma unroll
      for (int j = 0; j < 8; ++j)
        Ps[wave][(j + 8 * khalf) * 64 + nt * 16 + l16] = f32_to_bf16(s[nt][j]);

    Frag16 ap[2];
#pragma unroll
    for (int f = 0; f < 2; ++f)
#pragma unroll
      for (int j = 0; j < 8; ++j)
        ap[f].d[j] = *reinterpret_cast<const unsigned int*>(
            &Ps[wave][l16 * 64 + f * 32 + kmapA(j, khalf)]);

#pragma unroll
    for (int dt = 0; dt < 4; ++dt) {
      int dcol = dt * 16 + l16;
#pragma unroll
      for (int f = 0; f < 2; ++f) {
        Frag16 bv;
#pragma unroll
        for (int j = 0; j < 8; ++j) {
          int k = f * 32 + kmapB(j, khalf);
          bv.u[2 * j]     = Vs[k * 64 + dcol];
          bv.u[2 * j + 1] = Vs[(k + 1) * 64 + dcol];
        }
        oacc[dt] = wmma_bf16(ap[f], bv, oacc[dt]);
      }
    }
    __syncthreads();
  }

  // normalize + store bf16 [row][h*64+d]
#pragma unroll
  for (int dt = 0; dt < 4; ++dt)
#pragma unroll
    for (int j = 0; j < 8; ++j) {
      int m_l = j + 8 * khalf;
      size_t row_g = (size_t)b * T_SZ + q0 + qr + m_l;
      out[row_g * INNER + h * DHEAD + dt * 16 + l16] =
          f32_to_bf16(oacc[dt][j] / lrow[j]);
    }
}

// ---------------- host-side orchestration ----------------
extern "C" void kernel_launch(void* const* d_in, const int* in_sizes, int n_in,
                              void* d_out, int out_size, void* d_ws, size_t ws_size,
                              hipStream_t stream) {
  (void)in_sizes; (void)n_in; (void)out_size; (void)ws_size;
  const float* x     = (const float*)d_in[0];
  const float* ln1_g = (const float*)d_in[1];
  const float* ln1_b = (const float*)d_in[2];
  const float* w_qkv = (const float*)d_in[3];
  const float* w_out = (const float*)d_in[4];
  const float* b_out = (const float*)d_in[5];
  const float* ln2_g = (const float*)d_in[6];
  const float* ln2_b = (const float*)d_in[7];
  const float* w1    = (const float*)d_in[8];
  const float* b1    = (const float*)d_in[9];
  const float* w2    = (const float*)d_in[10];
  const float* b2    = (const float*)d_in[11];
  float* out = (float*)d_out;

  // workspace layout (74 MB peak, regions reused after their last consumer)
  char* ws = (char*)d_ws;
  const size_t MB = 1ull << 20;
  unsigned short* qkv_b  = (unsigned short*)(ws + 0);        // 24MB, later hid_b (32MB)
  unsigned short* hid_b  = (unsigned short*)(ws + 0);
  unsigned short* h_b    = (unsigned short*)(ws + 32 * MB);  // 8MB (h, then h2)
  unsigned short* wqkv_b = (unsigned short*)(ws + 40 * MB);  // 6MB, later w1_b (8MB)
  unsigned short* w1_b   = (unsigned short*)(ws + 40 * MB);
  unsigned short* attn_b = (unsigned short*)(ws + 48 * MB);  // 8MB, later w2_b (8MB)
  unsigned short* w2_b   = (unsigned short*)(ws + 48 * MB);
  float*          xmid   = (float*)(ws + 56 * MB);           // 16MB
  unsigned short* wout_b = (unsigned short*)(ws + 72 * MB);  // 2MB

  // weights -> bf16 (needed for step right below each)
  cvt_bf16_kernel<<<1024, 256, 0, stream>>>(w_qkv, wqkv_b, C_DIM * QKV_W);
  cvt_bf16_kernel<<<512, 256, 0, stream>>>(w_out, wout_b, INNER * C_DIM);

  // LN1 -> h (bf16)
  layernorm_bf16_kernel<<<BT, 256, 0, stream>>>(x, ln1_g, ln1_b, h_b);

  // QKV = h @ w_qkv  -> bf16
  gemm_bf16_kernel<0><<<dim3(QKV_W / 128, BT / 128), 256, 0, stream>>>(
      h_b, wqkv_b, BT, QKV_W, C_DIM, nullptr, nullptr, nullptr, qkv_b);

  // w1 -> bf16 (reuses wqkv slot; stream-ordered after QKV GEMM)
  cvt_bf16_kernel<<<1024, 256, 0, stream>>>(w1, w1_b, C_DIM * MLP_DIM);

  // causal flash attention -> bf16
  attention_kernel<<<dim3(T_SZ / 64, HEADS, B_SZ), 128, 0, stream>>>(qkv_b, attn_b);

  // x_mid = attn @ w_out + b_out + x   (fp32)
  gemm_bf16_kernel<1><<<dim3(C_DIM / 128, BT / 128), 256, 0, stream>>>(
      attn_b, wout_b, BT, C_DIM, INNER, b_out, x, xmid, nullptr);

  // w2 -> bf16 (reuses attn slot; stream-ordered after out-proj)
  cvt_bf16_kernel<<<1024, 256, 0, stream>>>(w2, w2_b, MLP_DIM * C_DIM);

  // LN2 -> h2 (bf16, reuses h slot)
  layernorm_bf16_kernel<<<BT, 256, 0, stream>>>(xmid, ln2_g, ln2_b, h_b);

  // hidden = gelu(h2 @ w1 + b1) -> bf16 (reuses qkv slot)
  gemm_bf16_kernel<2><<<dim3(MLP_DIM / 128, BT / 128), 256, 0, stream>>>(
      h_b, w1_b, BT, MLP_DIM, C_DIM, b1, nullptr, nullptr, hid_b);

  // out = hidden @ w2 + b2 + x_mid  (fp32)
  gemm_bf16_kernel<1><<<dim3(C_DIM / 128, BT / 128), 256, 0, stream>>>(
      hid_b, w2_b, BT, C_DIM, MLP_DIM, b2, xmid, out, nullptr);
}
